// Model_35003983463137
// MI455X (gfx1250) — matrix operand order repaired and checked by reading, not compiled, not run on hardware
//
#include <hip/hip_runtime.h>
#include <cstdint>

// ---------------------------------------------------------------------------
// HDC encode + classify for MI455X (gfx1250, wave32).
//
// Pipeline (all deterministic, all on `stream`):
//  1) pack_bits:    sign-bit pack level_weight (256x10000) and position
//                   (784x10000) into u32 words via wave32 ballot. 1.33 MB total
//                   -> whole working set is L2-resident (192 MB L2).
//  2) convert_w:    classify weights -> f16, n-major, zero-padded to [16][10240].
//  3) encode_part:  per (batch, p-chunk): XOR popcount with bit-sliced CSA
//                   counters (3-bit group counter per 7 p's -> 10-bit acc).
//                   32 d-values per lane-op; 32x less load traffic than f32.
//  4) encode_final: combine 4 partial counters, bit-sliced threshold
//                   (carry-out of m+632 <=> m>=392 <=> bundled<=0), emit
//                   enc as f16 {+1,-1,0-pad} [64][10240].
//  5) classify_wmma: logits = enc @ W^T via v_wmma_f32_16x16x32_f16.
//                   4 M-tiles x 8 K-chunks = 32 waves, LDS reduction.
// ---------------------------------------------------------------------------

typedef _Float16 v16h __attribute__((ext_vector_type(16)));
typedef _Float16 h8   __attribute__((ext_vector_type(8)));
typedef float    v8f  __attribute__((ext_vector_type(8)));

#define DIMS    10000
#define DPAD    10240          // 320 words * 32 bits
#define NWORDS  320
#define NPIX    784
#define NBATCH  64
#define NLVL    256
#define NCLS    10
#define PCHUNK  196            // 784 / 4
#define NGRP    28             // 196 / 7

// ---------------- 1) bit-pack L and P (bit = 1 iff value < 0) --------------
__global__ void pack_bits(const float* __restrict__ lw, const float* __restrict__ pos,
                          uint32_t* __restrict__ Lbits, uint32_t* __restrict__ Pbits) {
  uint32_t gid  = blockIdx.x * 256u + threadIdx.x;
  uint32_t wave = gid >> 5;
  uint32_t lane = gid & 31u;
  if (wave >= (NLVL + NPIX) * (uint32_t)NWORDS) return;   // wave-uniform
  uint32_t row = wave / NWORDS;
  uint32_t w   = wave % NWORDS;
  uint32_t d   = w * 32u + lane;
  const float* src;
  uint32_t* dst;
  if (row < NLVL) { src = lw  + (size_t)row * DIMS;          dst = Lbits + row * NWORDS + w; }
  else            { uint32_t pr = row - NLVL;
                    src = pos + (size_t)pr * DIMS;           dst = Pbits + pr  * NWORDS + w; }
  bool bit = (d < DIMS) ? (src[d] < 0.0f) : false;          // pad bits = 0
  uint32_t word = __builtin_amdgcn_ballot_w32(bit);         // coalesced 128B read per wave
  if (lane == 0) *dst = word;
}

// ---------------- 2) weights -> f16, n-major, zero padded ------------------
__global__ void convert_w(const float* __restrict__ cw, _Float16* __restrict__ wt) {
  int e = blockIdx.x * 256 + threadIdx.x;    // 16*10240 = 163840 elements
  int n = e / DPAD;
  int k = e % DPAD;
  float v = (n < NCLS && k < DIMS) ? cw[n * DIMS + k] : 0.0f;
  wt[e] = (_Float16)v;
}

// ---------------- 3) bind+bundle: bit-sliced XOR popcount ------------------
// grid = 64 batches * 4 p-chunks; block = 320 threads (one per 32-d word).
__global__ void encode_part(const float* __restrict__ x,
                            const uint32_t* __restrict__ Lbits,
                            const uint32_t* __restrict__ Pbits,
                            uint32_t* __restrict__ part) {
  __shared__ unsigned short sidx[PCHUNK];
  int b     = blockIdx.x >> 2;
  int chunk = blockIdx.x & 3;
  int pbase = chunk * PCHUNK;
  int tid   = threadIdx.x;
  if (tid < PCHUNK) {
    // idx = clip(round(x*255), 0, 255); rintf == round-half-even (matches jnp.round)
    float f = rintf(x[b * NPIX + pbase + tid] * 255.0f);
    f = fminf(fmaxf(f, 0.0f), 255.0f);
    sidx[tid] = (unsigned short)(int)f;
  }
  __syncthreads();

  int w = tid;                                   // word index 0..319
  uint32_t acc[10];                              // bit-sliced counter, cap 1023 >= 784
#pragma unroll
  for (int i = 0; i < 10; ++i) acc[i] = 0u;

  for (int g = 0; g < NGRP; ++g) {
    uint32_t c0 = 0, c1 = 0, c2 = 0;             // 3-bit group counter (7 adds max)
#pragma unroll
    for (int t = 0; t < 7; ++t) {
      int pl = g * 7 + t;
      uint32_t lwd = Lbits[(uint32_t)sidx[pl] * NWORDS + w];   // gathered row, coalesced in w
      uint32_t pwd = Pbits[(uint32_t)(pbase + pl) * NWORDS + w];
      uint32_t mm  = lwd ^ pwd;                  // bit=1 <=> product is -1
      uint32_t t0 = c0 & mm; c0 ^= mm;           // ripple add 1 bit, 3 levels
      uint32_t t1 = c1 & t0; c1 ^= t0;
      c2 ^= t1;
    }
    // acc += {c2,c1,c0}  (full adds at levels 0..2, then carry ripple)
    uint32_t carry = acc[0] & c0; acc[0] ^= c0;
    uint32_t s1 = acc[1] ^ c1; uint32_t co1 = (acc[1] & c1) | (carry & s1);
    acc[1] = s1 ^ carry; carry = co1;
    uint32_t s2 = acc[2] ^ c2; uint32_t co2 = (acc[2] & c2) | (carry & s2);
    acc[2] = s2 ^ carry; carry = co2;
#pragma unroll
    for (int i = 3; i < 10; ++i) { uint32_t tt = acc[i] & carry; acc[i] ^= carry; carry = tt; }
  }

  uint32_t base = (uint32_t)blockIdx.x * 10u * NWORDS + w;
#pragma unroll
  for (int i = 0; i < 10; ++i) part[base + i * NWORDS] = acc[i];
}

// ---------------- 4) combine partials, threshold, emit f16 enc -------------
__global__ void encode_final(const uint32_t* __restrict__ part,
                             _Float16* __restrict__ enc) {
  int b = blockIdx.x;
  int w = threadIdx.x;                           // 0..319
  uint32_t acc[10];
  uint32_t base0 = (uint32_t)(b * 4) * 10u * NWORDS + w;
#pragma unroll
  for (int i = 0; i < 10; ++i) acc[i] = part[base0 + i * NWORDS];
#pragma unroll
  for (int c = 1; c < 4; ++c) {                  // bit-sliced 10-bit adds
    uint32_t cb = (uint32_t)(b * 4 + c) * 10u * NWORDS + w;
    uint32_t carry = 0;
#pragma unroll
    for (int i = 0; i < 10; ++i) {
      uint32_t q  = part[cb + i * NWORDS];
      uint32_t s  = acc[i] ^ q;
      uint32_t co = (acc[i] & q) | (carry & s);
      acc[i] = s ^ carry; carry = co;
    }
  }
  // bundled = 784 - 2m;  bundled > 0 <=> m < 392.  carry-out of (m + 632) <=> m >= 392.
  uint32_t carry = 0;
#pragma unroll
  for (int i = 0; i < 10; ++i) {
    uint32_t bb  = ((632u >> i) & 1u) ? 0xFFFFFFFFu : 0u;
    uint32_t axb = acc[i] ^ bb;
    carry = (acc[i] & bb) | (carry & axb);
  }
  union { _Float16 h[32]; uint4 q[4]; } u;
  int dbase = w * 32;
#pragma unroll
  for (int j = 0; j < 32; ++j) {
    _Float16 hv = ((carry >> j) & 1u) ? (_Float16)(-1.0f) : (_Float16)(1.0f);
    if (dbase + j >= DIMS) hv = (_Float16)0.0f;  // zero pad -> contributes 0 to GEMM
    u.h[j] = hv;
  }
  uint4* dst = (uint4*)(enc + (size_t)b * DPAD + dbase);   // 64B-aligned
#pragma unroll
  for (int r = 0; r < 4; ++r) dst[r] = u.q[r];
}

// ---------------- 5) classify GEMM via WMMA f16->f32 -----------------------
// 1 block, 1024 threads = 32 waves: wave = kchunk(8) x mtile(4).
// Per-wave K loop = 40 steps of 32. ISA operand layouts:
//   A 16x32 f16 : lane->M(&15); elems 0..7 -> K = klo..klo+7, 8..15 -> K = 16+klo.. (klo = 8*(lane>=16))
//   B 32x16 f16 : lane->N(&15); same K order as A (weights stored n-major)
//   C/D 16x16 f32: elem r -> M = r + 8*(lane>=16), N = lane&15
__global__ void classify_wmma(const _Float16* __restrict__ enc,
                              const _Float16* __restrict__ wt,
                              float* __restrict__ out) {
  __shared__ float red[32 * 256];
  int tid   = threadIdx.x;
  int wid   = tid >> 5;
  int lane  = tid & 31;
  int mtile = wid & 3;
  int kc    = wid >> 2;
  int m     = lane & 15;
  int hi    = lane >> 4;
  int klo   = hi * 8;
  int koffB = hi * 8;

  v8f c = {};
  const _Float16* arow = enc + (size_t)(mtile * 16 + m) * DPAD;
  const _Float16* brow = wt  + (size_t)m * DPAD;          // N = lane&15 = m

#pragma unroll 4
  for (int kb = kc * 40; kb < kc * 40 + 40; ++kb) {
    int kbase = kb * 32;
    h8 alo = *(const h8*)(arow + kbase + klo);
    h8 ahi = *(const h8*)(arow + kbase + 16 + klo);
    h8 blo = *(const h8*)(brow + kbase + koffB);
    h8 bhi = *(const h8*)(brow + kbase + 16 + koffB);
    v16h a, bv;
#pragma unroll
    for (int j = 0; j < 8; ++j) {
      a[j] = alo[j]; a[j + 8] = ahi[j];
      bv[j] = blo[j]; bv[j + 8] = bhi[j];
    }
    c = __builtin_amdgcn_wmma_f32_16x16x32_f16(false, a, false, bv,
                                               (short)0, c, false, false);
  }
#pragma unroll
  for (int r = 0; r < 8; ++r)
    red[wid * 256 + (r + hi * 8) * 16 + m] = c[r];
  __syncthreads();

  if (tid < NBATCH * NCLS) {                     // 640 outputs
    int row = tid / NCLS, n = tid % NCLS;
    int mt = row >> 4, mm = row & 15;
    float s = 0.0f;
#pragma unroll
    for (int k = 0; k < 8; ++k) s += red[(k * 4 + mt) * 256 + mm * 16 + n];
    out[row * NCLS + n] = s;
  }
}

// ---------------------------------------------------------------------------
extern "C" void kernel_launch(void* const* d_in, const int* in_sizes, int n_in,
                              void* d_out, int out_size, void* d_ws, size_t ws_size,
                              hipStream_t stream) {
  (void)in_sizes; (void)n_in; (void)out_size; (void)ws_size;
  const float* x   = (const float*)d_in[0];   // [64,28,28]
  const float* pos = (const float*)d_in[1];   // [784,10000]
  const float* lw  = (const float*)d_in[2];   // [256,10000]
  const float* cw  = (const float*)d_in[3];   // [10,10000]
  float* out = (float*)d_out;                 // [64,10] f32

  char* ws = (char*)d_ws;                     // all offsets 256B-aligned
  uint32_t* Lbits = (uint32_t*)(ws + 0);            //  256*320*4 = 327,680 B
  uint32_t* Pbits = (uint32_t*)(ws + 327680);       //  784*320*4 = 1,003,520 B
  _Float16* WtT   = (_Float16*)(ws + 1331200);      //   16*10240*2 = 327,680 B
  _Float16* enc   = (_Float16*)(ws + 1658880);      //   64*10240*2 = 1,310,720 B
  uint32_t* part  = (uint32_t*)(ws + 2969600);      // 256*10*320*4 = 3,276,800 B
                                                    // total ~6.25 MB

  pack_bits   <<<41600, 256, 0, stream>>>(lw, pos, Lbits, Pbits);  // (256+784)*320 waves
  convert_w   <<<640,   256, 0, stream>>>(cw, WtT);
  encode_part <<<NBATCH * 4, NWORDS, 0, stream>>>(x, Lbits, Pbits, part);
  encode_final<<<NBATCH,     NWORDS, 0, stream>>>(part, enc);
  classify_wmma<<<1, 1024, 0, stream>>>(enc, WtT, out);
}
